// EmbeddingClassifier_55825984913811
// MI455X (gfx1250) — matrix operand order repaired and checked by reading, not compiled, not run on hardware
//
#include <hip/hip_runtime.h>
#include <hip/hip_bf16.h>

// Problem constants (match the JAX reference)
#define DD    256      // feature dim / K
#define MROWS 32768    // B*N = 4*8192
#define CCLS  1000     // classes (output cols)
#define CPAD  1024     // classes padded to multiple of 256
#define EPSF  1e-5f

typedef __attribute__((ext_vector_type(16))) __bf16 v16bf;
typedef __attribute__((ext_vector_type(8)))  __bf16 v8bf;
typedef __attribute__((ext_vector_type(8)))  float  v8f;

// ---------------------------------------------------------------------------
// Kernel 0: bulk f32 -> bf16 conversion (8 elements/thread, vectorized).
// n must be a multiple of 8 (true for all our uses).
// ---------------------------------------------------------------------------
__global__ void __launch_bounds__(256)
cvt_f32_to_bf16_kernel(const float* __restrict__ src,
                       __bf16* __restrict__ dst, int n) {
  int i = (blockIdx.x * 256 + threadIdx.x) * 8;
  if (i >= n) return;
  float4 u = *(const float4*)(src + i);
  float4 v = *(const float4*)(src + i + 4);
  v8bf o;
  o[0] = (__bf16)u.x; o[1] = (__bf16)u.y; o[2] = (__bf16)u.z; o[3] = (__bf16)u.w;
  o[4] = (__bf16)v.x; o[5] = (__bf16)v.y; o[6] = (__bf16)v.z; o[7] = (__bf16)v.w;
  *(v8bf*)(dst + i) = o;
}

// ---------------------------------------------------------------------------
// Kernel 1: protos = expmap0(embeddings) -> bf16 (rows >= CCLS zero-padded),
//           p_sq[c] = ||protos_c||^2.  One block of 256 threads per row.
// ---------------------------------------------------------------------------
__global__ void __launch_bounds__(256)
protos_expmap_kernel(const float* __restrict__ emb,
                     __bf16* __restrict__ protos,
                     float* __restrict__ p_sq) {
  int c = blockIdx.x;          // 0..CPAD-1
  int t = threadIdx.x;         // 0..255
  __shared__ float red[256];
  float e = (c < CCLS) ? emb[c * DD + t] : 0.f;
  red[t] = e * e;
  __syncthreads();
  #pragma unroll
  for (int s = 128; s > 0; s >>= 1) {
    if (t < s) red[t] += red[t + s];
    __syncthreads();
  }
  float sumsq = red[0];
  float nrm   = fmaxf(sqrtf(sumsq), EPSF);
  float scale = tanhf(nrm) / nrm;
  protos[c * DD + t] = (__bf16)(scale * e);
  if (t == 0) p_sq[c] = scale * scale * sumsq;
}

// ---------------------------------------------------------------------------
// Fragment loaders: bf16 row-major memory, lanes take K in contiguous order k = 16h + s for the WMMA
// 16x32 A / 32x16 B operands (one contiguous 16-element run per lane, same order in A and B).
// ---------------------------------------------------------------------------
__device__ __forceinline__ v16bf load_a_frag(const __bf16* __restrict__ arow,
                                             int k0, int half) {
  return *(const v16bf*)(arow + k0 + half * 16);         // K = k0+16h .. +15
}

__device__ __forceinline__ v16bf load_b_frag(const __bf16* __restrict__ bcol) {
  v8bf b0 = *(const v8bf*)(bcol);                        // K = k0+16h .. +15
  v8bf b1 = *(const v8bf*)(bcol + 8);
  return __builtin_shufflevector(b0, b1,
           0, 1, 2, 3, 4, 5, 6, 7, 8, 9, 10, 11, 12, 13, 14, 15);
}

// ---------------------------------------------------------------------------
// Kernel 2: y = features @ W^T + b  (bf16 WMMA, f32 accumulate), then
//           expmap0 per row (cross-wave LDS reduction for the row norm).
//           Emits x_mapped (bf16, row-major) and x_sq = ||x||^2.
// Block: 256 thr = 8 waves = 2 row-blocks(16) x 4 col-blocks(64) -> 32x256.
// ---------------------------------------------------------------------------
__global__ void __launch_bounds__(256)
proj_expmap_kernel(const __bf16* __restrict__ Af,    // features bf16 [MROWS, DD]
                   const __bf16* __restrict__ Wb,    // W bf16 [DD, DD] row-major
                   const float* __restrict__ bias,   // [DD]
                   __bf16* __restrict__ Xm,          // out [MROWS, DD] bf16
                   float* __restrict__ x_sq) {       // out [MROWS]
  __shared__ float lds_sum[32];
  __shared__ float lds_scale[32];
  const int tid = threadIdx.x;
  if (tid < 32) lds_sum[tid] = 0.f;
  __syncthreads();

  const int wave   = tid >> 5;
  const int lane   = tid & 31;
  const int half   = lane >> 4;   // 0: M rows 0-7 / K lo16, 1: M rows 8-15 / K hi16
  const int ln     = lane & 15;
  const int rowblk = wave & 1;
  const int colblk = wave >> 1;
  const int Mbase  = blockIdx.x * 32 + rowblk * 16;
  const int N0     = colblk * 64;

  v8f acc[4];
  #pragma unroll
  for (int t = 0; t < 4; ++t) acc[t] = v8f{};

  const __bf16* arow = Af + (size_t)(Mbase + ln) * DD;

  #pragma unroll
  for (int k0 = 0; k0 < DD; k0 += 32) {
    v16bf a = load_a_frag(arow, k0, half);
    #pragma unroll
    for (int t = 0; t < 4; ++t) {                // B-frag col n = W row n
      v16bf b = load_b_frag(Wb + (size_t)(N0 + t * 16 + ln) * DD
                               + k0 + half * 16);
      acc[t] = __builtin_amdgcn_wmma_f32_16x16x32_bf16(
                   false, a, false, b, (short)0, acc[t], false, false);
    }
  }

  // bias + per-row sum of squares (partial over this wave's 64 columns)
  float part[8];
  #pragma unroll
  for (int i = 0; i < 8; ++i) part[i] = 0.f;
  #pragma unroll
  for (int t = 0; t < 4; ++t) {
    float bv = bias[N0 + t * 16 + ln];
    #pragma unroll
    for (int i = 0; i < 8; ++i) {
      acc[t][i] += bv;
      part[i] = fmaf(acc[t][i], acc[t][i], part[i]);
    }
  }
  // reduce across the 16 lanes of the half-group, then across col-waves via LDS
  #pragma unroll
  for (int i = 0; i < 8; ++i) {
    float v = part[i];
    v += __shfl_xor(v, 1, 32);
    v += __shfl_xor(v, 2, 32);
    v += __shfl_xor(v, 4, 32);
    v += __shfl_xor(v, 8, 32);
    if (ln == 0) atomicAdd(&lds_sum[rowblk * 16 + half * 8 + i], v);
  }
  __syncthreads();
  if (tid < 32) {
    float ss  = lds_sum[tid];
    float nrm = fmaxf(sqrtf(ss), EPSF);
    float sc  = tanhf(nrm) / nrm;
    lds_scale[tid] = sc;
    x_sq[blockIdx.x * 32 + tid] = sc * sc * ss;
  }
  __syncthreads();

  // scale rows and emit bf16 x_mapped (GEMM2's A operand, pre-converted)
  #pragma unroll
  for (int t = 0; t < 4; ++t) {
    int n = N0 + t * 16 + ln;
    #pragma unroll
    for (int i = 0; i < 8; ++i) {
      int rl = rowblk * 16 + half * 8 + i;
      float sc = lds_scale[rl];
      Xm[(size_t)(blockIdx.x * 32 + rl) * DD + n] = (__bf16)(acc[t][i] * sc);
    }
  }
}

// ---------------------------------------------------------------------------
// Kernel 3: xp = x_mapped @ protos^T (bf16 WMMA) fused with the Poincare
//           distance epilogue.
// Wave tile 32(M) x 64(N): two A-frags share every B-frag -> 8 WMMA per
// k-step, halving per-output B-side L2 traffic vs a 16-row tile.
// Block: 8 waves = 2 row-waves(32) x 4 col-waves(64) -> 64 x 256 tile.
// Grid: (MROWS/64, CPAD/256)
// ---------------------------------------------------------------------------
__global__ void __launch_bounds__(256)
poincare_dist_kernel(const __bf16* __restrict__ Xm,   // [MROWS, DD]
                     const __bf16* __restrict__ P,    // [CPAD, DD]
                     const float* __restrict__ x_sq,  // [MROWS]
                     const float* __restrict__ p_sq,  // [CPAD]
                     const float* __restrict__ logit_scale, // [1]
                     float* __restrict__ out) {       // [MROWS, CCLS]
  const int tid    = threadIdx.x;
  const int wave   = tid >> 5;
  const int lane   = tid & 31;
  const int half   = lane >> 4;
  const int ln     = lane & 15;
  const int rowblk = wave & 1;
  const int colblk = wave >> 1;
  const int Mbase  = blockIdx.x * 64 + rowblk * 32;   // this wave: rows [Mbase, Mbase+32)
  const int Nsup   = blockIdx.y * 256 + colblk * 64;

  v8f acc[2][4];
  #pragma unroll
  for (int r = 0; r < 2; ++r)
    #pragma unroll
    for (int t = 0; t < 4; ++t) acc[r][t] = v8f{};

  const __bf16* arow0 = Xm + (size_t)(Mbase + ln) * DD;
  const __bf16* arow1 = Xm + (size_t)(Mbase + 16 + ln) * DD;

  #pragma unroll
  for (int k0 = 0; k0 < DD; k0 += 32) {
    v16bf a0 = load_a_frag(arow0, k0, half);
    v16bf a1 = load_a_frag(arow1, k0, half);
    #pragma unroll
    for (int t = 0; t < 4; ++t) {
      v16bf b = load_b_frag(P + (size_t)(Nsup + t * 16 + ln) * DD
                              + k0 + half * 16);
      acc[0][t] = __builtin_amdgcn_wmma_f32_16x16x32_bf16(
                      false, a0, false, b, (short)0, acc[0][t], false, false);
      acc[1][t] = __builtin_amdgcn_wmma_f32_16x16x32_bf16(
                      false, a1, false, b, (short)0, acc[1][t], false, false);
    }
  }

  const float lsc = fminf(__expf(logit_scale[0]), 100.f);
  float xs[2][8];
  #pragma unroll
  for (int r = 0; r < 2; ++r)
    #pragma unroll
    for (int i = 0; i < 8; ++i)
      xs[r][i] = x_sq[Mbase + r * 16 + half * 8 + i];

  #pragma unroll
  for (int t = 0; t < 4; ++t) {
    int n = Nsup + t * 16 + ln;
    float ps  = p_sq[n];
    float omp = 1.f - ps;
    #pragma unroll
    for (int r = 0; r < 2; ++r) {
      #pragma unroll
      for (int i = 0; i < 8; ++i) {
        float xq    = xs[r][i];
        float diff  = fmaxf(xq + ps - 2.f * acc[r][t][i], 0.f);
        float denom = fmaxf((1.f - xq) * omp, EPSF);
        float arg   = fmaxf(1.f + 2.f * diff / denom, 1.f + EPSF);
        float dist  = acoshf(arg);
        if (n < CCLS) {
          out[(size_t)(Mbase + r * 16 + half * 8 + i) * CCLS + n] = -dist * lsc;
        }
      }
    }
  }
}

// ---------------------------------------------------------------------------
// Host-side launch.  Inputs: features, W, b, embeddings, logit_scale.
// Workspace layout:
//   [x_mapped bf16 16MB][protos bf16 512KB][x_sq 128KB][p_sq 4KB]
//   [features bf16 16.78MB][W bf16 128KB]
// ---------------------------------------------------------------------------
extern "C" void kernel_launch(void* const* d_in, const int* in_sizes, int n_in,
                              void* d_out, int out_size, void* d_ws, size_t ws_size,
                              hipStream_t stream) {
  const float* features    = (const float*)d_in[0];
  const float* W           = (const float*)d_in[1];
  const float* bias        = (const float*)d_in[2];
  const float* emb         = (const float*)d_in[3];
  const float* logit_scale = (const float*)d_in[4];
  float* out = (float*)d_out;

  char* ws = (char*)d_ws;
  const size_t XM_BYTES   = (size_t)MROWS * DD * sizeof(__bf16);  // 16 MB
  const size_t P_BYTES    = (size_t)CPAD  * DD * sizeof(__bf16);  // 512 KB
  const size_t XSQ_BYTES  = (size_t)MROWS * sizeof(float);        // 128 KB
  const size_t PSQ_BYTES  = (size_t)CPAD  * sizeof(float);        // 4 KB
  const size_t AF_BYTES   = (size_t)MROWS * DD * sizeof(__bf16);  // 16 MB
  size_t off = 0;
  __bf16* x_mapped = (__bf16*)(ws + off); off += XM_BYTES;
  __bf16* protos   = (__bf16*)(ws + off); off += P_BYTES;
  float*  x_sq     = (float*)(ws + off);  off += XSQ_BYTES;
  float*  p_sq     = (float*)(ws + off);  off += PSQ_BYTES;
  __bf16* feat_bf  = (__bf16*)(ws + off); off += AF_BYTES;
  __bf16* w_bf     = (__bf16*)(ws + off);

  const int NFEAT = MROWS * DD;   // 8,388,608
  const int NW    = DD * DD;      // 65,536

  // 0) one-shot f32 -> bf16 operand conversions (hoisted out of GEMM loops)
  cvt_f32_to_bf16_kernel<<<NFEAT / (256 * 8), 256, 0, stream>>>(features,
                                                                feat_bf, NFEAT);
  cvt_f32_to_bf16_kernel<<<NW / (256 * 8), 256, 0, stream>>>(W, w_bf, NW);
  // 1) prototypes -> Poincare ball (tiny)
  protos_expmap_kernel<<<CPAD, 256, 0, stream>>>(emb, protos, p_sq);
  // 2) projection GEMM + expmap0 (bf16 WMMA)
  proj_expmap_kernel<<<MROWS / 32, 256, 0, stream>>>(feat_bf, w_bf, bias,
                                                     x_mapped, x_sq);
  // 3) big GEMM + Poincare distance epilogue (bf16 WMMA, 32x64 wave tile)
  poincare_dist_kernel<<<dim3(MROWS / 64, CPAD / 256), 256, 0, stream>>>(
      x_mapped, protos, x_sq, p_sq, logit_scale, out);
}
